// FinalASLModel_51917564674694
// MI455X (gfx1250) — matrix operand order repaired and checked by reading, not compiled, not run on hardware
//
#include <hip/hip_runtime.h>
#include <math.h>

// ---------------------------------------------------------------------------
// CDNA5 / gfx1250 fused attention+FFN block.
//  - All GEMMs + global attention (QK^T, PV) on v_wmma_f32_16x16x32_bf16.
//  - GEMM: 128x64x32 tiles, double-buffered LDS staging via
//    global_load_async_to_lds_b128 (ASYNCcnt) when available.
//  - V produced pre-transposed (VT[b,h][d][s]) by its projection GEMM ->
//    attention PV B-fragments are contiguous global loads (no LDS transpose).
//  - Softmax scales folded into the Q projection epilogues.
//  - bf16 activations between GEMMs; fp32 only for residual + LayerNorm.
// ---------------------------------------------------------------------------

typedef unsigned short u16;
typedef unsigned int   u32;
typedef __attribute__((ext_vector_type(8)))  unsigned short us8;
typedef __attribute__((ext_vector_type(16))) __bf16         v16bf;
typedef __attribute__((ext_vector_type(2)))  __bf16         v2bf;
typedef __attribute__((ext_vector_type(8)))  float          v8f;
typedef __attribute__((__vector_size__(4 * sizeof(int)))) int v4i;

#if defined(__gfx1250__) && __has_builtin(__builtin_amdgcn_global_load_async_to_lds_b128)
#define HAVE_ASYNC_LDS 1
typedef __attribute__((address_space(1))) v4i as1_v4i;   // global (AS1)
typedef __attribute__((address_space(3))) v4i as3_v4i;   // LDS (AS3)
#if __has_builtin(__builtin_amdgcn_s_wait_asynccnt)
#define WAIT_ASYNC(n) __builtin_amdgcn_s_wait_asynccnt(n)
#else
#define WAIT_ASYNC(n) asm volatile("s_wait_asynccnt %0" ::"i"(n) : "memory")
#endif
#else
#define HAVE_ASYNC_LDS 0
#endif

union FragU { v16bf v; us8 h[2]; };

static __device__ __forceinline__ u16 f2bf(float f) {
  u32 u = __float_as_uint(f);
  u += 0x7FFFu + ((u >> 16) & 1u);  // round-to-nearest-even
  return (u16)(u >> 16);
}
// Packed pair convert: returns {lo=bf16(a), hi=bf16(b)} as u32.
static __device__ __forceinline__ u32 f2bf_pk(float a, float b) {
#if defined(__gfx1250__) && __has_builtin(__builtin_amdgcn_cvt_pk_bf16_f32)
  union { v2bf v; u32 u; } cv;
  cv.v = __builtin_amdgcn_cvt_pk_bf16_f32(a, b);
  return cv.u;
#else
  return (u32)f2bf(a) | ((u32)f2bf(b) << 16);
#endif
}
static __device__ __forceinline__ float bf2f(u16 b) {
  return __uint_as_float(((u32)b) << 16);
}
static __device__ __forceinline__ v8f zero8() {
  v8f z = {0.f, 0.f, 0.f, 0.f, 0.f, 0.f, 0.f, 0.f};
  return z;
}

// A-fragment (16x32 bf16, row-major src, row stride ld elements).
static __device__ __forceinline__ v16bf frag_a(const u16* src, int ld, int lane) {
  const int m = lane & 15, hf = lane >> 4;
  const u16* p = src + (size_t)m * ld + hf * 8;
  FragU f;
  f.h[0] = *(const us8*)(p);
  f.h[1] = *(const us8*)(p + 16);
  return f.v;
}
// B-fragment (32x16 bf16) from K-major storage src[n][k], stride ld.
static __device__ __forceinline__ v16bf frag_b(const u16* src, int ld, int lane) {
  const int n = lane & 15, hf = lane >> 4;
  const u16* p = src + (size_t)n * ld + hf * 8;
  FragU f;
  f.h[0] = *(const us8*)(p);
  f.h[1] = *(const us8*)(p + 16);
  return f.v;
}
static __device__ __forceinline__ v8f wmma_bf16(v16bf a, v16bf b, v8f c) {
  return __builtin_amdgcn_wmma_f32_16x16x32_bf16(false, a, false, b,
                                                 (short)0, c, false, false);
}

// ---------------------------------------------------------------------------
__global__ void transpose_w_kernel(const float* __restrict__ w,
                                   u16* __restrict__ wt, int K, int N) {
  int idx = blockIdx.x * 256 + threadIdx.x;
  if (idx >= K * N) return;
  int k = idx / N, n = idx - k * N;
  wt[(size_t)n * K + k] = f2bf(w[idx]);
}

__global__ void convert_bf16_kernel(const float* __restrict__ in,
                                    u16* __restrict__ out, int n) {
  int idx = blockIdx.x * 256 + threadIdx.x;
  if (idx < n) out[idx] = f2bf(in[idx]);
}

// ---------------------------------------------------------------------------
// Tiled WMMA GEMM: out = act( (A[M,K](bf16) @ Bt[N,K]^T(bf16) + bias)*scale )
// 256 threads = 8 waves (4 m x 2 n), tile 128x64x32, double-buffered LDS.
// ACT: 0=none, 1=SiLU.   SCALED: multiply (acc+bias) by `scale`.
// OUT_MODE: 0 = fp32, 1 = bf16 (ldOut/colOff), 2 = bf16 transposed attention-V
//           layout VT[(b*8+h)*64+d][s] (hardcoded S=1024, H=8, hd=64).
// ---------------------------------------------------------------------------
template <int ACT, int OUT_MODE, int SCALED>
__global__ __launch_bounds__(256) void gemm_bf16_kernel(
    const u16* __restrict__ A, const u16* __restrict__ Bt,
    const float* __restrict__ bias, void* __restrict__ outp,
    int M, int N, int K, int ldA, int ldOut, int colOff, float scale) {
  __shared__ u16 As[2][128 * 40];  // K-tile 32, row stride 40 (16B aligned)
  __shared__ u16 Bs[2][64 * 40];

  const int tid  = threadIdx.x;
  const int lane = tid & 31;
  const int wv   = tid >> 5;
  const int mW   = wv & 3, nW = wv >> 2;
  const int mBase = blockIdx.y * 128;
  const int nBase = blockIdx.x * 64;

  // Stage one 128x32 A tile + 64x32 B tile into LDS buffer `buf`.
  auto stage = [&](int buf, int k0) {
#if HAVE_ASYNC_LDS
#pragma unroll
    for (int it = 0; it < 2; ++it) {
      int id = tid + it * 256, row = id >> 2, c = id & 3;
      __builtin_amdgcn_global_load_async_to_lds_b128(
          (as1_v4i*)(void*)(A + (size_t)(mBase + row) * ldA + k0 + c * 8),
          (as3_v4i*)&As[buf][row * 40 + c * 8], 0, 0);
    }
    {
      int row = tid >> 2, c = tid & 3;
      __builtin_amdgcn_global_load_async_to_lds_b128(
          (as1_v4i*)(void*)(Bt + (size_t)(nBase + row) * K + k0 + c * 8),
          (as3_v4i*)&Bs[buf][row * 40 + c * 8], 0, 0);
    }
#else
#pragma unroll
    for (int it = 0; it < 2; ++it) {
      int id = tid + it * 256, row = id >> 2, c = id & 3;
      *(us8*)&As[buf][row * 40 + c * 8] =
          *(const us8*)(A + (size_t)(mBase + row) * ldA + k0 + c * 8);
    }
    {
      int row = tid >> 2, c = tid & 3;
      *(us8*)&Bs[buf][row * 40 + c * 8] =
          *(const us8*)(Bt + (size_t)(nBase + row) * K + k0 + c * 8);
    }
#endif
  };

  v8f acc[2][2];
#pragma unroll
  for (int i = 0; i < 2; ++i)
#pragma unroll
    for (int j = 0; j < 2; ++j) acc[i][j] = zero8();

  const int nk = K >> 5;
  stage(0, 0);
  for (int kt = 0; kt < nk; ++kt) {
    const int cur = kt & 1;
#if HAVE_ASYNC_LDS
    if (kt + 1 < nk) {
      stage(cur ^ 1, (kt + 1) * 32);  // prefetch next tile into other buffer
      WAIT_ASYNC(3);                  // stage(kt) complete (in-order retire)
    } else {
      WAIT_ASYNC(0);
    }
#else
    if (kt + 1 < nk) stage(cur ^ 1, (kt + 1) * 32);
#endif
    __syncthreads();

    v16bf aF[2], bF[2];
#pragma unroll
    for (int i = 0; i < 2; ++i)
      aF[i] = frag_a(&As[cur][(mW * 32 + i * 16) * 40], 40, lane);
#pragma unroll
    for (int j = 0; j < 2; ++j)
      bF[j] = frag_b(&Bs[cur][(nW * 32 + j * 16) * 40], 40, lane);
#pragma unroll
    for (int i = 0; i < 2; ++i)
#pragma unroll
      for (int j = 0; j < 2; ++j) acc[i][j] = wmma_bf16(aF[i], bF[j], acc[i][j]);
    __syncthreads();
  }

  // ---- epilogue ----
  const int hf = lane >> 4, nn = lane & 15;
#pragma unroll
  for (int i = 0; i < 2; ++i) {
#pragma unroll
    for (int j = 0; j < 2; ++j) {
      const int col = nBase + nW * 32 + j * 16 + nn;
      const float bv = bias[col];
#pragma unroll
      for (int r = 0; r < 8; r += 2) {
        const int rowg = mBase + mW * 32 + i * 16 + hf * 8 + r;
        float v0 = acc[i][j][r] + bv;
        float v1 = acc[i][j][r + 1] + bv;
        if (SCALED) { v0 *= scale; v1 *= scale; }
        if (ACT == 1) {  // SiLU
          v0 = v0 / (1.f + __expf(-v0));
          v1 = v1 / (1.f + __expf(-v1));
        }
        if (OUT_MODE == 0) {
          ((float*)outp)[(size_t)rowg * ldOut + colOff + col] = v0;
          ((float*)outp)[(size_t)(rowg + 1) * ldOut + colOff + col] = v1;
        } else if (OUT_MODE == 1) {
          const u32 pk = f2bf_pk(v0, v1);
          ((u16*)outp)[(size_t)rowg * ldOut + colOff + col] = (u16)pk;
          ((u16*)outp)[(size_t)(rowg + 1) * ldOut + colOff + col] = (u16)(pk >> 16);
        } else {  // VT[(b*8+h)*64+d][s]; r,r+1 are adjacent s -> one u32 store
          const int bb = rowg >> 10, s = rowg & 1023;
          const int h = col >> 6, dd = col & 63;
          *(u32*)&((u16*)outp)[(((size_t)(bb * 8 + h)) * 64 + dd) * 1024 + s] =
              f2bf_pk(v0, v1);
        }
      }
    }
  }
}

// ---------------------------------------------------------------------------
// Global attention, flash-style. 4 waves/block, one 16-row Q tile per wave.
// Q arrives pre-scaled by 1/sqrt(hd); V pre-transposed (VT[bh][d][s]).
// Only LDS use is the wave-private P (C->A relayout) + bpermute reductions.
// ---------------------------------------------------------------------------
__global__ __launch_bounds__(128) void attn_global_kernel(
    const u16* __restrict__ qg, const u16* __restrict__ kg,
    const u16* __restrict__ vt, u16* __restrict__ og) {
  __shared__ u16 Ps[4][16 * 40];
  const int wid = threadIdx.x >> 5;
  const int lane = threadIdx.x & 31;
  const int qt = blockIdx.x * 4 + wid;  // 0..63 (S/16)
  const int bh = blockIdx.y;            // 0..63 (B*H)
  const int b = bh >> 3, h = bh & 7;
  const int tok0 = b * 1024;
  const int colh = h * 64;
  const int hf = lane >> 4, nn = lane & 15;
  u16* ps = &Ps[wid][0];
  const u16* vbase = vt + (size_t)bh * 64 * 1024;

  v16bf qf[2];
#pragma unroll
  for (int c = 0; c < 2; ++c)
    qf[c] = frag_a(qg + (size_t)(tok0 + qt * 16) * 512 + colh + c * 32, 512, lane);

  v8f ov[4];
#pragma unroll
  for (int j = 0; j < 4; ++j) ov[j] = zero8();
  float rm[8], rl[8];
#pragma unroll
  for (int r = 0; r < 8; ++r) { rm[r] = -1e30f; rl[r] = 0.f; }

  for (int kv = 0; kv < 1024; kv += 32) {
    // ---- scores: two 16x16 tiles, contraction hd=64 (2 chunks) ----
    v8f sc[2];
    sc[0] = zero8(); sc[1] = zero8();
#pragma unroll
    for (int j = 0; j < 2; ++j)
#pragma unroll
      for (int c = 0; c < 2; ++c) {
        v16bf kf = frag_b(kg + (size_t)(tok0 + kv + j * 16) * 512 + colh + c * 32,
                          512, lane);
        sc[j] = wmma_bf16(qf[c], kf, sc[j]);
      }
    // ---- online softmax (Q pre-scaled; scores already scaled) ----
#pragma unroll
    for (int r = 0; r < 8; ++r) {
      float s0 = sc[0][r];
      float s1 = sc[1][r];
      float mx = fmaxf(s0, s1);
#pragma unroll
      for (int off = 1; off < 16; off <<= 1) mx = fmaxf(mx, __shfl_xor(mx, off, 32));
      float mnew = fmaxf(rm[r], mx);
      float corr = __expf(rm[r] - mnew);
      float p0 = __expf(s0 - mnew);
      float p1 = __expf(s1 - mnew);
      float pssum = p0 + p1;
#pragma unroll
      for (int off = 1; off < 16; off <<= 1) pssum += __shfl_xor(pssum, off, 32);
      rl[r] = rl[r] * corr + pssum;
      rm[r] = mnew;
#pragma unroll
      for (int j = 0; j < 4; ++j) ov[j][r] = ov[j][r] * corr;
      const u32 pk = f2bf_pk(p0, p1);
      ps[(hf * 8 + r) * 40 + nn]      = (u16)pk;
      ps[(hf * 8 + r) * 40 + 16 + nn] = (u16)(pk >> 16);
    }
    // Wave-private LDS RAW: DS ops retire in order per wave; fence the
    // compiler and drain DScnt before the A-fragment reads.
    asm volatile("s_wait_dscnt 0x0" ::: "memory");
    // ---- PV: O(16x64) += P(16x32) @ V(32x64), V direct from global VT ----
    v16bf pf = frag_a(ps, 40, lane);
#pragma unroll
    for (int j = 0; j < 4; ++j) {
      v16bf vb = frag_b(vbase + (size_t)(j * 16) * 1024 + kv, 1024, lane);
      ov[j] = wmma_bf16(pf, vb, ov[j]);
    }
    asm volatile("" ::: "memory");  // keep next iter's P stores after reads
  }
  // ---- normalize & store ----
#pragma unroll
  for (int j = 0; j < 4; ++j)
#pragma unroll
    for (int r = 0; r < 8; r += 2) {
      const int rowg = tok0 + qt * 16 + hf * 8 + r;
      const u32 pk = f2bf_pk(ov[j][r] / rl[r], ov[j][r + 1] / rl[r + 1]);
      og[(size_t)rowg * 512 + colh + j * 16 + nn] = (u16)pk;
      og[(size_t)(rowg + 1) * 512 + colh + j * 16 + nn] = (u16)(pk >> 16);
    }
}

// ---------------------------------------------------------------------------
// Local windowed attention: HT=4 heads, hdt=128, W=5 (tiny FLOPs -> VALU).
// Q arrives pre-scaled by 1/sqrt(hdt).
// ---------------------------------------------------------------------------
__global__ __launch_bounds__(128) void attn_local_kernel(
    const u16* __restrict__ qg, const u16* __restrict__ kg,
    const u16* __restrict__ vg, u16* __restrict__ og) {
  const int t = blockIdx.x;  // token
  const int s = t & 1023;
  const int base = t - s;
  const int h = threadIdx.x >> 5;
  const int lane = threadIdx.x & 31;
  const int col = h * 128 + lane;

  float qd[4];
#pragma unroll
  for (int i = 0; i < 4; ++i) qd[i] = bf2f(qg[(size_t)t * 512 + col + i * 32]);

  float sc[5];
#pragma unroll
  for (int w = 0; w < 5; ++w) {
    int ns = s + w - 2;
    float p = -1e30f;
    if (ns >= 0 && ns < 1024) {  // uniform across the wave
      const size_t rk = (size_t)(base + ns) * 512 + col;
      float a = 0.f;
#pragma unroll
      for (int i = 0; i < 4; ++i) a += qd[i] * bf2f(kg[rk + i * 32]);
#pragma unroll
      for (int off = 16; off > 0; off >>= 1) a += __shfl_xor(a, off, 32);
      p = a;  // scale folded into Q projection
    }
    sc[w] = p;
  }
  float m = sc[0];
#pragma unroll
  for (int w = 1; w < 5; ++w) m = fmaxf(m, sc[w]);
  float pe[5], sum = 0.f;
#pragma unroll
  for (int w = 0; w < 5; ++w) {
    pe[w] = (sc[w] > -1e29f) ? __expf(sc[w] - m) : 0.f;
    sum += pe[w];
  }
  const float isum = 1.f / sum;
#pragma unroll
  for (int i = 0; i < 4; ++i) {
    float a = 0.f;
#pragma unroll
    for (int w = 0; w < 5; ++w) {
      int ns = s + w - 2;
      ns = ns < 0 ? 0 : (ns > 1023 ? 1023 : ns);  // clipped; pe==0 masks
      a += pe[w] * bf2f(vg[(size_t)(base + ns) * 512 + col + i * 32]);
    }
    og[(size_t)t * 512 + col + i * 32] = f2bf(a * isum);
  }
}

// ---------------------------------------------------------------------------
// Residual + LayerNorm over D=512; fp32 out + optional bf16 copy.
// ---------------------------------------------------------------------------
__global__ __launch_bounds__(256) void ln_kernel(
    const float* __restrict__ xa, const float* __restrict__ xr,
    const float* __restrict__ g, const float* __restrict__ bta,
    float* __restrict__ outF, u16* __restrict__ outB) {
  __shared__ float s1[256], s2[256];
  const int row = blockIdx.x, t = threadIdx.x;
  const size_t base = (size_t)row * 512;
  float v0 = xa[base + t] + xr[base + t];
  float v1 = xa[base + t + 256] + xr[base + t + 256];
  s1[t] = v0 + v1;
  s2[t] = v0 * v0 + v1 * v1;
  __syncthreads();
  for (int o = 128; o > 0; o >>= 1) {
    if (t < o) { s1[t] += s1[t + o]; s2[t] += s2[t + o]; }
    __syncthreads();
  }
  const float mean = s1[0] * (1.f / 512.f);
  const float var = s2[0] * (1.f / 512.f) - mean * mean;
  const float inv = rsqrtf(var + 1e-5f);
  const float o0 = (v0 - mean) * inv * g[t] + bta[t];
  const float o1 = (v1 - mean) * inv * g[t + 256] + bta[t + 256];
  outF[base + t] = o0;
  outF[base + t + 256] = o1;
  if (outB) {
    const u32 pk = f2bf_pk(o0, o1);
    outB[base + t] = (u16)pk;
    outB[base + t + 256] = (u16)(pk >> 16);
  }
}

// ---------------------------------------------------------------------------
// Host orchestration
// ---------------------------------------------------------------------------
extern "C" void kernel_launch(void* const* d_in, const int* in_sizes, int n_in,
                              void* d_out, int out_size, void* d_ws, size_t ws_size,
                              hipStream_t stream) {
  (void)in_sizes; (void)n_in; (void)out_size; (void)ws_size;

  const float* x      = (const float*)d_in[0];
  const float* g_wq   = (const float*)d_in[1];
  const float* g_wk   = (const float*)d_in[2];
  const float* g_wv   = (const float*)d_in[3];
  const float* g_bq   = (const float*)d_in[4];
  const float* g_bk   = (const float*)d_in[5];
  const float* g_bv   = (const float*)d_in[6];
  const float* g_wo   = (const float*)d_in[7];
  const float* g_bo   = (const float*)d_in[8];
  const float* t_wq   = (const float*)d_in[9];
  const float* t_wk   = (const float*)d_in[10];
  const float* t_wv   = (const float*)d_in[11];
  const float* t_bq   = (const float*)d_in[12];
  const float* t_bk   = (const float*)d_in[13];
  const float* t_bv   = (const float*)d_in[14];
  const float* t_wo   = (const float*)d_in[15];
  const float* t_bo   = (const float*)d_in[16];
  const float* fus_w1 = (const float*)d_in[17];
  const float* fus_b1 = (const float*)d_in[18];
  const float* fus_w2 = (const float*)d_in[19];
  const float* fus_b2 = (const float*)d_in[20];
  const float* ffn_w1 = (const float*)d_in[21];
  const float* ffn_b1 = (const float*)d_in[22];
  const float* ffn_w2 = (const float*)d_in[23];
  const float* ffn_b2 = (const float*)d_in[24];
  const float* gn_g   = (const float*)d_in[25];
  const float* gn_b   = (const float*)d_in[26];
  const float* fn_g   = (const float*)d_in[27];
  const float* fn_b   = (const float*)d_in[28];

  const int NT = 8 * 1024;  // tokens
  char* ws = (char*)d_ws;
  const size_t KB1 = 1024, MB1 = KB1 * KB1;
  // Weights (bf16, [N][K]) — persistent
  u16* GWQ = (u16*)(ws + 0 * MB1);
  u16* GWK = (u16*)(ws + 0 * MB1 + 512 * KB1);
  u16* GWV = (u16*)(ws + 1 * MB1);
  u16* GWO = (u16*)(ws + 1 * MB1 + 512 * KB1);
  u16* TWQ = (u16*)(ws + 2 * MB1);
  u16* TWK = (u16*)(ws + 2 * MB1 + 512 * KB1);
  u16* TWV = (u16*)(ws + 3 * MB1);
  u16* TWO = (u16*)(ws + 3 * MB1 + 512 * KB1);
  u16* FW1 = (u16*)(ws + 4 * MB1);   // 2 MB
  u16* FW2 = (u16*)(ws + 6 * MB1);   // 1 MB
  u16* FFW1 = (u16*)(ws + 7 * MB1);  // 1 MB
  u16* FFW2 = (u16*)(ws + 8 * MB1);  // 1 MB
  // Activations (regions reused once dead); peak = 81 MB
  u16*  XB  = (u16*)(ws + 9 * MB1);    // x bf16          [9,17)
  u16*  QB  = (u16*)(ws + 17 * MB1);   // q / ql          [17,25)
  u16*  KBF = (u16*)(ws + 25 * MB1);   // k / kl          [25,33)
  u16*  VB  = (u16*)(ws + 33 * MB1);   // VT / vl         [33,41)
  u16*  AO  = (u16*)(ws + 41 * MB1);   // attn out bf16   [41,49)
  u16*  XC  = (u16*)(ws + 49 * MB1);   // concat bf16     [49,65)
  u16*  H1  = (u16*)(ws + 65 * MB1);   // fus hidden      [65,81)
  float* XF = (float*)(ws + 17 * MB1); // fus out fp32    [17,33) reuse q/k
  float* X2 = (float*)(ws + 33 * MB1); // LN1 fp32        [33,49) reuse v/ao
  u16*  X2B = (u16*)(ws + 49 * MB1);   // LN1 bf16        [49,57) reuse xc
  u16*  FFH = (u16*)(ws + 57 * MB1);   // ffn hidden      [57,73) reuse xc/h1
  float* XFF = (float*)(ws + 17 * MB1);// ffn out fp32    [17,33) reuse xf

  auto T = [&](const float* w, u16* wt, int K, int N) {
    int n = K * N;
    transpose_w_kernel<<<(n + 255) / 256, 256, 0, stream>>>(w, wt, K, N);
  };
  // mode: 0=fp32, 1=bf16, 2=bf16 transposed V, 3=bf16 scaled
  auto G = [&](const u16* A, const u16* Bt, const float* bias, void* out,
               int M, int N, int K, int ldA, int ldOut, int colOff,
               int act, int mode, float scale) {
    dim3 grid(N / 64, M / 128);
    if (mode == 3)
      gemm_bf16_kernel<0, 1, 1><<<grid, 256, 0, stream>>>(A, Bt, bias, out, M, N, K, ldA, ldOut, colOff, scale);
    else if (mode == 2)
      gemm_bf16_kernel<0, 2, 0><<<grid, 256, 0, stream>>>(A, Bt, bias, out, M, N, K, ldA, ldOut, colOff, 1.f);
    else if (mode == 0)
      gemm_bf16_kernel<0, 0, 0><<<grid, 256, 0, stream>>>(A, Bt, bias, out, M, N, K, ldA, ldOut, colOff, 1.f);
    else if (act == 1)
      gemm_bf16_kernel<1, 1, 0><<<grid, 256, 0, stream>>>(A, Bt, bias, out, M, N, K, ldA, ldOut, colOff, 1.f);
    else
      gemm_bf16_kernel<0, 1, 0><<<grid, 256, 0, stream>>>(A, Bt, bias, out, M, N, K, ldA, ldOut, colOff, 1.f);
  };

  // 1) weights -> bf16 transposed
  T(g_wq, GWQ, 512, 512);   T(g_wk, GWK, 512, 512);
  T(g_wv, GWV, 512, 512);   T(g_wo, GWO, 512, 512);
  T(t_wq, TWQ, 512, 512);   T(t_wk, TWK, 512, 512);
  T(t_wv, TWV, 512, 512);   T(t_wo, TWO, 512, 512);
  T(fus_w1, FW1, 1024, 1024); T(fus_w2, FW2, 1024, 512);
  T(ffn_w1, FFW1, 512, 1024); T(ffn_w2, FFW2, 1024, 512);

  // 2) x -> bf16
  convert_bf16_kernel<<<(NT * 512 + 255) / 256, 256, 0, stream>>>(x, XB, NT * 512);

  // 3) global QKV projections (Q pre-scaled by 1/sqrt(64); V transposed)
  G(XB, GWQ, g_bq, QB,  NT, 512, 512, 512, 512, 0, 0, 3, 0.125f);
  G(XB, GWK, g_bk, KBF, NT, 512, 512, 512, 512, 0, 0, 1, 1.f);
  G(XB, GWV, g_bv, VB,  NT, 512, 512, 512, 0,   0, 0, 2, 1.f);

  // 4) global attention (flash, WMMA)
  attn_global_kernel<<<dim3(16, 64), 128, 0, stream>>>(QB, KBF, VB, AO);

  // 5) xg = attn @ g_wo + g_bo -> xc[:, 0:512]
  G(AO, GWO, g_bo, XC, NT, 512, 512, 512, 1024, 0, 0, 1, 1.f);

  // 6) local QKV projections (Q pre-scaled by 1/sqrt(128))
  G(XB, TWQ, t_bq, QB,  NT, 512, 512, 512, 512, 0, 0, 3, 0.08838834764831845f);
  G(XB, TWK, t_bk, KBF, NT, 512, 512, 512, 512, 0, 0, 1, 1.f);
  G(XB, TWV, t_bv, VB,  NT, 512, 512, 512, 512, 0, 0, 1, 1.f);

  // 7) local windowed attention (W=5)
  attn_local_kernel<<<NT, 128, 0, stream>>>(QB, KBF, VB, AO);

  // 8) xl = attn_l @ t_wo + t_bo -> xc[:, 512:1024]
  G(AO, TWO, t_bo, XC, NT, 512, 512, 512, 1024, 512, 0, 1, 1.f);

  // 9) h1 = silu(xc @ fus_w1 + fus_b1)
  G(XC, FW1, fus_b1, H1, NT, 1024, 1024, 1024, 1024, 0, 1, 1, 1.f);

  // 10) xf = h1 @ fus_w2 + fus_b2 (fp32)
  G(H1, FW2, fus_b2, XF, NT, 512, 1024, 1024, 512, 0, 0, 0, 1.f);

  // 11) x2 = LN(x + xf); also bf16 copy
  ln_kernel<<<NT, 256, 0, stream>>>(x, XF, gn_g, gn_b, X2, X2B);

  // 12) ffh = silu(x2 @ ffn_w1 + ffn_b1)
  G(X2B, FFW1, ffn_b1, FFH, NT, 1024, 512, 512, 1024, 0, 1, 1, 1.f);

  // 13) xff = ffh @ ffn_w2 + ffn_b2 (fp32)
  G(FFH, FFW2, ffn_b2, XFF, NT, 512, 1024, 1024, 512, 0, 0, 0, 1.f);

  // 14) out = LN(x2 + xff)
  ln_kernel<<<NT, 256, 0, stream>>>(X2, XFF, fn_g, fn_b, (float*)d_out, nullptr);
}
